// SleepTransformer_50294067036773
// MI455X (gfx1250) — matrix operand order repaired and checked by reading, not compiled, not run on hardware
//
#include <hip/hip_runtime.h>
#include <hip/hip_bf16.h>

// ---------------- problem constants (from reference) ----------------
#define NLAYER 2
#define DMODEL 512
#define NHEAD  8
#define DHEAD  64
#define DFFN   2048
#define NCLS   5
#define WINR   10
#define NBATCH 32
#define SEQT   1024
#define MTOK   (NBATCH * SEQT)   // 32768 rows

typedef __attribute__((ext_vector_type(16))) _Float16 v16h;
typedef __attribute__((ext_vector_type(8)))  _Float16 v8h;
typedef __attribute__((ext_vector_type(8)))  float    v8f;

// ---------------- f32 -> f16 conversion ----------------
__global__ __launch_bounds__(256) void cvt_f16_kernel(const float* __restrict__ in,
                                                      _Float16* __restrict__ out, int n) {
    int i = blockIdx.x * blockDim.x + threadIdx.x;
    if (i < n) out[i] = (_Float16)in[i];
}

// ---------------- LayerNorm: one wave32 per row, f32 in -> f16 out ----------------
__global__ __launch_bounds__(256) void ln_f16_kernel(const float* __restrict__ x,
                                                     const float* __restrict__ g,
                                                     const float* __restrict__ b,
                                                     _Float16* __restrict__ y) {
    int row  = blockIdx.x * 8 + threadIdx.y;
    int lane = threadIdx.x;
    const float* xr = x + (size_t)row * DMODEL;
    float s = 0.f, s2 = 0.f;
    #pragma unroll
    for (int i = lane; i < DMODEL; i += 32) { float v = xr[i]; s += v; s2 += v * v; }
    #pragma unroll
    for (int m = 16; m >= 1; m >>= 1) { s += __shfl_xor(s, m, 32); s2 += __shfl_xor(s2, m, 32); }
    float mean = s * (1.0f / DMODEL);
    float var  = s2 * (1.0f / DMODEL) - mean * mean;
    float rs   = rsqrtf(var + 1e-5f);
    _Float16* yr = y + (size_t)row * DMODEL;
    #pragma unroll
    for (int i = lane; i < DMODEL; i += 32)
        yr[i] = (_Float16)((xr[i] - mean) * rs * g[i] + b[i]);
}

// ---------------- WMMA GEMM: Y[M,N] = epi(A[M,K](f16) @ W[N,K](f16)^T + bias) ----------------
// Block = 4 waves; block tile = 32(M) x 256(N); each wave owns a 32x64 strip:
// 2 A fragments x 4 N tiles = 8 f32 accumulators (64 VGPRs). Each B fragment now feeds
// TWO wmma ops, halving W-load issue per matrix op. The 32xK A slab is shared by all
// 4 waves -> staged through LDS in 128-wide K chunks (8.5 KB of 320 KB/WGP), padded row
// stride (136 halves = 272B) so 16-lane ds_load_b128 fragment reads are bank-conflict-free.
// W streams from global/L2 (weights are L2-resident on the 192MB L2); one global_prefetch
// per W row one K-stage ahead.
// Fragment layouts used here (A/B k = 16g + s is this code's order, not the ISA table order):
//   A 16x32 f16 : lane l -> row l%16 ; group g=l/16: slot s -> k = g*16 + s (contiguous 16 halves)
//   B 32x16 f16 : lane l -> column l%16 ; group g -> k = g*16 + s (contiguous 16 halves)
//   C/D 16x16 f32: vgpr v, lane l -> m = v + 8*g, n = l%16
#define BM        32
#define BK        128
#define ATILE_LD  136   // padded LDS row stride in halves

template<int RELU, int RESID, int OUTF16, int OUTF32>
__global__ __launch_bounds__(128) void gemm_wmma_kernel(
    const _Float16* __restrict__ A,     // [M,K]
    const _Float16* __restrict__ W,     // [N,K] row-major (W^T is the Kx16 B operand)
    const float*    __restrict__ bias,  // [N]
    const float*    __restrict__ resid, // [M,N] fp32 residual (may alias outF)
    float*          __restrict__ outF,  // [M,N] fp32
    _Float16*       __restrict__ outH,  // [M,N] f16
    int N, int K)
{
    __shared__ _Float16 aTile[BM * ATILE_LD];   // 8.5 KB

    const int tid   = threadIdx.x;
    const int lane  = tid & 31;
    const int wave  = tid >> 5;
    const int m0    = blockIdx.x * BM;
    const int nbase = blockIdx.y * 256 + wave * 64;
    const int lm    = lane & 15;
    const int g     = lane >> 4;

    v8f acc[2][4] = {};

    // cooperative A staging: 128 threads x 32 halves (64B) cover 32x128 halves
    const int arow_id = tid >> 2;            // 0..31
    const int acol    = (tid & 3) * 32;      // 0,32,64,96
    const _Float16* aptr = A + (size_t)(m0 + arow_id) * K + acol;

    const _Float16* wrowb[4];
    #pragma unroll
    for (int t = 0; t < 4; ++t)
        wrowb[t] = W + (size_t)(nbase + t * 16 + lm) * K + g * 16;

    for (int k0 = 0; k0 < K; k0 += BK) {
        // ---- stage 32xBK A slab into LDS ----
        v16h av0 = *(const v16h*)(aptr + k0);
        v16h av1 = *(const v16h*)(aptr + k0 + 16);
        *(v16h*)(&aTile[arow_id * ATILE_LD + acol])      = av0;
        *(v16h*)(&aTile[arow_id * ATILE_LD + acol + 16]) = av1;
        __syncthreads();

        // prefetch W rows one K-stage ahead (global_prefetch_b8; speculative OOB is dropped)
        #pragma unroll
        for (int t = 0; t < 4; ++t)
            __builtin_prefetch(wrowb[t] + k0 + BK, 0, 1);

        #pragma unroll
        for (int kk = 0; kk < BK; kk += 32) {
            v16h af0, af1;
            {   // A fragment for rows m0 + 0..15
                v8h alo = *(const v8h*)(&aTile[lm * ATILE_LD + kk + g * 16]);
                v8h ahi = *(const v8h*)(&aTile[lm * ATILE_LD + kk + g * 16 + 8]);
                #pragma unroll
                for (int i = 0; i < 8; ++i) { af0[i] = alo[i]; af0[i + 8] = ahi[i]; }
            }
            {   // A fragment for rows m0 + 16..31
                v8h alo = *(const v8h*)(&aTile[(16 + lm) * ATILE_LD + kk + g * 16]);
                v8h ahi = *(const v8h*)(&aTile[(16 + lm) * ATILE_LD + kk + g * 16 + 8]);
                #pragma unroll
                for (int i = 0; i < 8; ++i) { af1[i] = alo[i]; af1[i + 8] = ahi[i]; }
            }
            #pragma unroll
            for (int t = 0; t < 4; ++t) {
                v16h bf = *(const v16h*)(wrowb[t] + k0 + kk);   // one B frag -> two wmma
                acc[0][t] = __builtin_amdgcn_wmma_f32_16x16x32_f16(
                                false, af0, false, bf, (short)0, acc[0][t], false, false);
                acc[1][t] = __builtin_amdgcn_wmma_f32_16x16x32_f16(
                                false, af1, false, bf, (short)0, acc[1][t], false, false);
            }
        }
        __syncthreads();
    }

    #pragma unroll
    for (int mt = 0; mt < 2; ++mt) {
        #pragma unroll
        for (int t = 0; t < 4; ++t) {
            int n = nbase + t * 16 + lm;
            float bv = bias[n];
            #pragma unroll
            for (int v = 0; v < 8; ++v) {
                int m = m0 + mt * 16 + v + 8 * g;
                size_t idx = (size_t)m * N + n;
                float val = acc[mt][t][v] + bv;
                if (RESID)  val += resid[idx];
                if (RELU)   val = val > 0.f ? val : 0.f;
                if (OUTF32) outF[idx] = val;
                if (OUTF16) outH[idx] = (_Float16)val;
            }
        }
    }
}

// ---------------- banded local attention, one wave32 per (b,h,t) query ----------------
// qkv: [MTOK, 3*DMODEL] f16 ; o: [MTOK, DMODEL] f16
__global__ __launch_bounds__(256) void attn_local_kernel(
    const _Float16* __restrict__ qkv,
    const unsigned char* __restrict__ pad,   // [B*T] bool, true = padded key
    _Float16* __restrict__ o)
{
    const int lane = threadIdx.x;
    const int t    = blockIdx.x * 8 + threadIdx.y;
    const int b    = blockIdx.y / NHEAD;
    const int h    = blockIdx.y % NHEAD;
    const int row  = b * SEQT + t;

    const _Float16* q = qkv + (size_t)row * (3 * DMODEL) + h * DHEAD;
    int jlo = t - WINR; if (jlo < 0) jlo = 0;
    int jhi = t + WINR; if (jhi > SEQT - 1) jhi = SEQT - 1;
    const int nk = jhi - jlo + 1;   // 11..21

    float score = -3.0e38f;
    if (lane < nk) {
        int j = jlo + lane;
        const _Float16* kp = qkv + (size_t)(b * SEQT + j) * (3 * DMODEL) + DMODEL + h * DHEAD;
        float s = 0.f;
        #pragma unroll
        for (int d = 0; d < DHEAD; ++d) s += (float)q[d] * (float)kp[d];
        score = s * 0.125f + (pad[b * SEQT + j] ? -1e9f : 0.0f);  // 1/sqrt(64)=0.125
    }
    float mx = score;
    #pragma unroll
    for (int m = 16; m >= 1; m >>= 1) mx = fmaxf(mx, __shfl_xor(mx, m, 32));
    float e = (lane < nk) ? __expf(score - mx) : 0.0f;
    float ssum = e;
    #pragma unroll
    for (int m = 16; m >= 1; m >>= 1) ssum += __shfl_xor(ssum, m, 32);
    float a = e / ssum;

    const int d0 = lane * 2;                 // each lane owns 2 of the 64 dims
    float o0 = 0.f, o1 = 0.f;
    for (int jj = 0; jj < nk; ++jj) {
        float aj = __shfl(a, jj, 32);
        const _Float16* vp = qkv + (size_t)(b * SEQT + jlo + jj) * (3 * DMODEL)
                             + 2 * DMODEL + h * DHEAD;
        o0 += aj * (float)vp[d0];
        o1 += aj * (float)vp[d0 + 1];
    }
    _Float16* orow = o + (size_t)row * DMODEL + h * DHEAD;
    orow[d0]     = (_Float16)o0;
    orow[d0 + 1] = (_Float16)o1;
}

// ---------------- classifier: out[m,c] = h[m,:] . Wc[c,:] + bc[c]  (N=5, fp32) ----------------
__global__ __launch_bounds__(256) void classifier_kernel(
    const float* __restrict__ h, const float* __restrict__ Wc,
    const float* __restrict__ bc, float* __restrict__ out)
{
    int i = blockIdx.x * blockDim.x + threadIdx.x;
    if (i >= MTOK * NCLS) return;
    int m = i / NCLS, c = i % NCLS;
    const float* hr = h  + (size_t)m * DMODEL;
    const float* wr = Wc + (size_t)c * DMODEL;
    float s = bc[c];
    #pragma unroll 8
    for (int d = 0; d < DMODEL; ++d) s += hr[d] * wr[d];
    out[i] = s;
}

// ---------------- host orchestration ----------------
extern "C" void kernel_launch(void* const* d_in, const int* in_sizes, int n_in,
                              void* d_out, int out_size, void* d_ws, size_t ws_size,
                              hipStream_t stream) {
    (void)in_sizes; (void)n_in; (void)out_size; (void)ws_size;
    const float*         x    = (const float*)d_in[0];
    const unsigned char* pad  = (const unsigned char*)d_in[1];
    const float* Wqkv = (const float*)d_in[2];
    const float* bqkv = (const float*)d_in[3];
    const float* Wo   = (const float*)d_in[4];
    const float* bo   = (const float*)d_in[5];
    const float* ln1g = (const float*)d_in[6];
    const float* ln1b = (const float*)d_in[7];
    const float* W1   = (const float*)d_in[8];
    const float* b1   = (const float*)d_in[9];
    const float* W2   = (const float*)d_in[10];
    const float* b2   = (const float*)d_in[11];
    const float* ln2g = (const float*)d_in[12];
    const float* ln2b = (const float*)d_in[13];
    const float* Wc   = (const float*)d_in[14];
    const float* bc   = (const float*)d_in[15];

    char* wsb = (char*)d_ws;
    size_t off = 0;
    auto alloc = [&](size_t bytes) -> void* {
        void* p = wsb + off;
        off += (bytes + 255) & ~(size_t)255;
        return p;
    };
    float*     hbuf   = (float*)    alloc((size_t)MTOK * DMODEL * 4);            // 64 MB
    _Float16*  ybuf   = (_Float16*) alloc((size_t)MTOK * DMODEL * 2);            // 32 MB
    _Float16*  bigbuf = (_Float16*) alloc((size_t)MTOK * DFFN   * 2);            // 128 MB (qkv or z)
    _Float16*  obuf   = (_Float16*) alloc((size_t)MTOK * DMODEL * 2);            // 32 MB
    _Float16*  wqkv16 = (_Float16*) alloc((size_t)NLAYER * 3 * DMODEL * DMODEL * 2);
    _Float16*  wo16   = (_Float16*) alloc((size_t)NLAYER * DMODEL * DMODEL * 2);
    _Float16*  w116   = (_Float16*) alloc((size_t)NLAYER * DFFN * DMODEL * 2);
    _Float16*  w216   = (_Float16*) alloc((size_t)NLAYER * DMODEL * DFFN * 2);

    // weight conversion to f16
    {
        int n;
        n = NLAYER * 3 * DMODEL * DMODEL;
        cvt_f16_kernel<<<(n + 255) / 256, 256, 0, stream>>>(Wqkv, wqkv16, n);
        n = NLAYER * DMODEL * DMODEL;
        cvt_f16_kernel<<<(n + 255) / 256, 256, 0, stream>>>(Wo, wo16, n);
        n = NLAYER * DFFN * DMODEL;
        cvt_f16_kernel<<<(n + 255) / 256, 256, 0, stream>>>(W1, w116, n);
        n = NLAYER * DMODEL * DFFN;
        cvt_f16_kernel<<<(n + 255) / 256, 256, 0, stream>>>(W2, w216, n);
    }

    // residual stream starts as x (fp32)
    hipMemcpyAsync(hbuf, x, (size_t)MTOK * DMODEL * 4, hipMemcpyDeviceToDevice, stream);

    const dim3 lnGrid(MTOK / 8), lnBlk(32, 8);
    const dim3 attnGrid(SEQT / 8, NBATCH * NHEAD), attnBlk(32, 8);

    for (int l = 0; l < NLAYER; ++l) {
        // ---- attention sublayer ----
        ln_f16_kernel<<<lnGrid, lnBlk, 0, stream>>>(hbuf, ln1g + l * DMODEL, ln1b + l * DMODEL, ybuf);
        // qkv = y @ Wqkv^T + bqkv  -> f16  [MTOK, 1536]
        gemm_wmma_kernel<0, 0, 1, 0><<<dim3(MTOK / BM, (3 * DMODEL) / 256), 128, 0, stream>>>(
            ybuf, wqkv16 + (size_t)l * 3 * DMODEL * DMODEL, bqkv + (size_t)l * 3 * DMODEL,
            nullptr, nullptr, bigbuf, 3 * DMODEL, DMODEL);
        attn_local_kernel<<<attnGrid, attnBlk, 0, stream>>>(bigbuf, pad, obuf);
        // h = h + o @ Wo^T + bo  -> fp32
        gemm_wmma_kernel<0, 1, 0, 1><<<dim3(MTOK / BM, DMODEL / 256), 128, 0, stream>>>(
            obuf, wo16 + (size_t)l * DMODEL * DMODEL, bo + (size_t)l * DMODEL,
            hbuf, hbuf, nullptr, DMODEL, DMODEL);
        // ---- FFN sublayer ----
        ln_f16_kernel<<<lnGrid, lnBlk, 0, stream>>>(hbuf, ln2g + l * DMODEL, ln2b + l * DMODEL, ybuf);
        // z = relu(y @ W1^T + b1) -> f16 [MTOK, 2048]
        gemm_wmma_kernel<1, 0, 1, 0><<<dim3(MTOK / BM, DFFN / 256), 128, 0, stream>>>(
            ybuf, w116 + (size_t)l * DFFN * DMODEL, b1 + (size_t)l * DFFN,
            nullptr, nullptr, bigbuf, DFFN, DMODEL);
        // h = h + z @ W2^T + b2 -> fp32
        gemm_wmma_kernel<0, 1, 0, 1><<<dim3(MTOK / BM, DMODEL / 256), 128, 0, stream>>>(
            bigbuf, w216 + (size_t)l * DMODEL * DFFN, b2 + (size_t)l * DMODEL,
            hbuf, hbuf, nullptr, DMODEL, DFFN);
    }

    classifier_kernel<<<(MTOK * NCLS + 255) / 256, 256, 0, stream>>>(hbuf, Wc, bc, (float*)d_out);
}
